// TLSTM_Enc_60455959659036
// MI455X (gfx1250) — matrix operand order repaired and checked by reading, not compiled, not run on hardware
//
#include <hip/hip_runtime.h>
#include <hip/hip_bf16.h>

// ---------------------------------------------------------------------------
// Time-aware LSTM (2 layers) for MI455X / gfx1250.
// bf16 WMMA (v_wmma_f32_16x16x32_bf16) GEMMs, f32 state carry, and
// double-buffered GLOBAL_LOAD_ASYNC_TO_LDS_B128 tile staging (ASYNCcnt).
// Per step: kernel A computes h_adj = h + (T-1)*tanh(h@Wd^T+bd)  (bf16 out),
//           kernel B computes z = [x|h_adj]@W^T + b and the LSTM gates fused.
// ---------------------------------------------------------------------------

typedef __bf16 bf16_t;
typedef __bf16 v8bf  __attribute__((ext_vector_type(8)));
typedef __bf16 v16bf __attribute__((ext_vector_type(16)));
typedef float  v8f   __attribute__((ext_vector_type(8)));

#define B_DIM   512
#define SEQ_LEN 256
#define H_DIM   512
#define ECONSTF 2.7183f
#define LDSS    40   // LDS row stride in bf16 elems: 32 + 8 pad (80B rows, 16B aligned)

static __device__ __forceinline__ float sigf(float x) {
  return 1.0f / (1.0f + __expf(-x));
}
static __device__ __forceinline__ bf16_t to_bf16(float x) {
  return (bf16_t)x;
}

// Async copy of 16 bytes global -> LDS via the gfx1250 async-DMA path.
// LDS destination = low 32 bits of the generic pointer (aperture rule).
static __device__ __forceinline__ void async_ld16(const bf16_t* g, bf16_t* l) {
  unsigned loff = (unsigned)(unsigned long long)(void*)l;
  asm volatile("global_load_async_to_lds_b128 %0, %1, off"
               :: "v"(loff), "v"((unsigned long long)g)
               : "memory");
}
static __device__ __forceinline__ void wait_async0() {
#if __has_builtin(__builtin_amdgcn_s_wait_asynccnt)
  __builtin_amdgcn_s_wait_asynccnt(0);
#else
  asm volatile("s_wait_asynccnt 0x0" ::: "memory");
#endif
}

// A-fragment (16x32 bf16): lane(0-15)=row M, kh=lane>>4 selects K-half.
// elements 0..7 = K kh*8..kh*8+7 ; elements 8..15 = K 16+kh*8..
static __device__ __forceinline__ v16bf load_frag_a(const bf16_t* row, int kh) {
  v8bf lo = *(const v8bf*)(row + kh * 8);
  v8bf hi = *(const v8bf*)(row + kh * 8 + 16);
  return __builtin_shufflevector(lo, hi, 0,1,2,3,4,5,6,7,8,9,10,11,12,13,14,15);
}
// B-fragment (32x16 bf16): lane(0-15)=col N, elements 0..7 = K kh*8..kh*8+7 ; elements 8..15 = K 16+kh*8..
static __device__ __forceinline__ v16bf load_frag_b(const bf16_t* row, int kh) {
  v8bf lo = *(const v8bf*)(row + kh * 8);
  v8bf hi = *(const v8bf*)(row + kh * 8 + 16);
  return __builtin_shufflevector(lo, hi, 0,1,2,3,4,5,6,7,8,9,10,11,12,13,14,15);
}

// ---------------------------------------------------------------------------
// Kernel A: h_adj = h_f32 + (T-1) * tanh(h_bf16 @ Wd^T + bd),  T = 1/log(dt+e)
// GEMM M=512(B) N=512(H) K=512.  Block: 128 thr (4 waves), tile 64(M) x 32(N).
// ---------------------------------------------------------------------------
__global__ __launch_bounds__(128) void tlstm_hadj_kernel(
    const bf16_t* __restrict__ Hb,    // [512][512] bf16 h
    const bf16_t* __restrict__ Wd,    // [512][512] bf16
    const float*  __restrict__ bd,    // [512]
    const float*  __restrict__ Hf32,  // [512][512] f32 h
    const float*  __restrict__ DT,    // &inputs_t[0][j][63]
    long dtStride,                    // SEQ*64
    bf16_t* __restrict__ Hadj)        // [512][512] bf16 out
{
  __shared__ __align__(16) bf16_t Asm[2 * 64 * LDSS];
  __shared__ __align__(16) bf16_t Bsm[2 * 32 * LDSS];

  const int tid   = threadIdx.x;
  const int wave  = tid >> 5;
  const int lane  = tid & 31;
  const int nl    = lane & 15;
  const int kh    = lane >> 4;
  const int mBase = blockIdx.y * 64;
  const int nBase = blockIdx.x * 32;

  v8f acc[2] = {};

  auto issue = [&](int k0, int buf) {
    bf16_t* Ab = &Asm[buf * 64 * LDSS];
    bf16_t* Bb = &Bsm[buf * 32 * LDSS];
#pragma unroll
    for (int i = 0; i < 2; ++i) {                 // A tile: 64 rows x 32 k
      int c = tid + i * 128;
      int r = c >> 2, kk = (c & 3) * 8;
      async_ld16(&Hb[(long)(mBase + r) * H_DIM + k0 + kk], &Ab[r * LDSS + kk]);
    }
    {                                             // B tile: 32 rows(n) x 32 k
      int r = tid >> 2, kk = (tid & 3) * 8;
      async_ld16(&Wd[(long)(nBase + r) * H_DIM + k0 + kk], &Bb[r * LDSS + kk]);
    }
  };

  issue(0, 0);
  wait_async0();
  __syncthreads();

  for (int k0 = 0, it = 0; k0 < H_DIM; k0 += 32, ++it) {
    const int cur = it & 1;
    if (k0 + 32 < H_DIM) issue(k0 + 32, cur ^ 1);   // prefetch next tile

    const bf16_t* Ab = &Asm[cur * 64 * LDSS];
    const bf16_t* Bb = &Bsm[cur * 32 * LDSS];
    v16bf afrag = load_frag_a(&Ab[(wave * 16 + nl) * LDSS], kh);
#pragma unroll
    for (int t = 0; t < 2; ++t) {
      v16bf bfrag = load_frag_b(&Bb[(t * 16 + nl) * LDSS], kh);
      acc[t] = __builtin_amdgcn_wmma_f32_16x16x32_bf16(
          false, afrag, false, bfrag, (short)0, acc[t], false, false);
    }

    wait_async0();
    __syncthreads();
  }

#pragma unroll
  for (int t = 0; t < 2; ++t) {
    const int n = nBase + t * 16 + nl;
    const float bdn = bd[n];
#pragma unroll
    for (int r = 0; r < 8; ++r) {
      const int m = mBase + wave * 16 + kh * 8 + r;
      const float dt = DT[(long)m * dtStride];
      const float T  = 1.0f / __logf(dt + ECONSTF);
      const float cst = tanhf(acc[t][r] + bdn);
      const long idx = (long)m * H_DIM + n;
      Hadj[idx] = to_bf16(Hf32[idx] + (T - 1.0f) * cst);
    }
  }
}

// ---------------------------------------------------------------------------
// Kernel B: z = [x | h_adj] @ W^T + b ; fused gates.
// GEMM M=512(B) N=2048(4 gates x 512) K=Kx+512.
// Block: 128 thr, tile 64(M) x 32(N per gate) x 4 gates (A-frag reused 4x).
// ---------------------------------------------------------------------------
__global__ __launch_bounds__(128) void tlstm_gates_kernel(
    const bf16_t* __restrict__ Ax, long ldax, int Kx,  // x part (bf16, padded)
    const bf16_t* __restrict__ Ah,                     // h_adj [512][512]
    const bf16_t* __restrict__ Wx,                     // [2048][Kx]
    const bf16_t* __restrict__ Wh,                     // [2048][512]
    const float*  __restrict__ bias,                   // [2048]
    float*  __restrict__ Cst,                          // c state [512][512]
    float*  __restrict__ Hf32,                         // h out f32
    bf16_t* __restrict__ Hb16,                         // h out bf16 [512][512]
    bf16_t* __restrict__ Lout, long ldlout)            // optional layer-out slot
{
  __shared__ __align__(16) bf16_t Asm[2 * 64 * LDSS];
  __shared__ __align__(16) bf16_t Bsm[2 * 128 * LDSS];

  const int tid   = threadIdx.x;
  const int wave  = tid >> 5;
  const int lane  = tid & 31;
  const int nl    = lane & 15;
  const int kh    = lane >> 4;
  const int mBase = blockIdx.y * 64;
  const int nBase = blockIdx.x * 32;

  v8f acc[4][2] = {};

  auto issue = [&](int k0, int buf) {
    const bool inx   = (k0 < Kx);
    const bf16_t* as = inx ? (Ax + k0) : (Ah + (k0 - Kx));
    const long   lda = inx ? ldax : (long)H_DIM;
    const bf16_t* ws = inx ? (Wx + k0) : (Wh + (k0 - Kx));
    const long   ldw = inx ? (long)Kx : (long)H_DIM;
    bf16_t* Ab = &Asm[buf * 64 * LDSS];
    bf16_t* Bb = &Bsm[buf * 128 * LDSS];
#pragma unroll
    for (int i = 0; i < 2; ++i) {                 // A tile: 64 x 32
      int c = tid + i * 128;
      int r = c >> 2, kk = (c & 3) * 8;
      async_ld16(&as[(long)(mBase + r) * lda + kk], &Ab[r * LDSS + kk]);
    }
#pragma unroll
    for (int i = 0; i < 4; ++i) {                 // B tile: 128 rows (4g x 32n) x 32
      int c = tid + i * 128;
      int r = c >> 2, kk = (c & 3) * 8;
      int g = r >> 5, nloc = r & 31;
      async_ld16(&ws[(long)(g * H_DIM + nBase + nloc) * ldw + kk],
                 &Bb[r * LDSS + kk]);
    }
  };

  const int Ktot = Kx + H_DIM;

  issue(0, 0);
  wait_async0();
  __syncthreads();

  for (int k0 = 0, it = 0; k0 < Ktot; k0 += 32, ++it) {
    const int cur = it & 1;
    if (k0 + 32 < Ktot) issue(k0 + 32, cur ^ 1);  // prefetch next tile

    const bf16_t* Ab = &Asm[cur * 64 * LDSS];
    const bf16_t* Bb = &Bsm[cur * 128 * LDSS];
    v16bf afrag = load_frag_a(&Ab[(wave * 16 + nl) * LDSS], kh);
#pragma unroll
    for (int g = 0; g < 4; ++g) {
#pragma unroll
      for (int t = 0; t < 2; ++t) {
        v16bf bfrag = load_frag_b(&Bb[(g * 32 + t * 16 + nl) * LDSS], kh);
        acc[g][t] = __builtin_amdgcn_wmma_f32_16x16x32_bf16(
            false, afrag, false, bfrag, (short)0, acc[g][t], false, false);
      }
    }

    wait_async0();
    __syncthreads();
  }

  // Fused LSTM gate epilogue: order ci, cf, co, cg.
#pragma unroll
  for (int t = 0; t < 2; ++t) {
    const int n = nBase + t * 16 + nl;
    const float bi = bias[n];
    const float bf = bias[H_DIM + n];
    const float bo = bias[2 * H_DIM + n];
    const float bg = bias[3 * H_DIM + n];
#pragma unroll
    for (int r = 0; r < 8; ++r) {
      const int m = mBase + wave * 16 + kh * 8 + r;
      const long idx = (long)m * H_DIM + n;
      const float zi = acc[0][t][r] + bi;
      const float zf = acc[1][t][r] + bf;
      const float zo = acc[2][t][r] + bo;
      const float zg = acc[3][t][r] + bg;
      const float cp = Cst[idx];
      const float cn = sigf(zf) * cp + sigf(zi) * tanhf(zg);
      const float hn = sigf(zo) * tanhf(cn);
      Cst[idx]  = cn;
      Hf32[idx] = hn;
      Hb16[idx] = to_bf16(hn);
      if (Lout) Lout[(long)m * ldlout + n] = to_bf16(hn);
    }
  }
}

// ---------------------------------------------------------------------------
// Prep: f32 -> bf16 slice conversion with zero padding beyond copy_cols.
// ---------------------------------------------------------------------------
__global__ void conv_f32_bf16_kernel(bf16_t* __restrict__ dst,
                                     const float* __restrict__ src,
                                     long rows, long src_ld, long src_off,
                                     int copy_cols, int dst_cols) {
  long i = (long)blockIdx.x * blockDim.x + threadIdx.x;
  long total = rows * (long)dst_cols;
  if (i >= total) return;
  long r = i / dst_cols;
  int  c = (int)(i - r * dst_cols);
  float v = (c < copy_cols) ? src[r * src_ld + src_off + c] : 0.0f;
  dst[i] = to_bf16(v);
}

__global__ void zero_u4_kernel(uint4* __restrict__ p, long n4) {
  long i = (long)blockIdx.x * blockDim.x + threadIdx.x;
  if (i < n4) p[i] = make_uint4(0u, 0u, 0u, 0u);
}

// ---------------------------------------------------------------------------
extern "C" void kernel_launch(void* const* d_in, const int* in_sizes, int n_in,
                              void* d_out, int out_size, void* d_ws, size_t ws_size,
                              hipStream_t stream) {
  (void)in_sizes; (void)n_in; (void)out_size; (void)ws_size;

  const float* in_t = (const float*)d_in[0];   // [512][256][64]
  const float* W0   = (const float*)d_in[1];   // [2048][575]
  const float* b0   = (const float*)d_in[2];   // [2048]
  const float* Wd0  = (const float*)d_in[3];   // [512][512]
  const float* bd0  = (const float*)d_in[4];   // [512]
  const float* W1   = (const float*)d_in[5];   // [2048][1024]
  const float* b1   = (const float*)d_in[6];
  const float* Wd1  = (const float*)d_in[7];
  const float* bd1  = (const float*)d_in[8];

  // ---- carve workspace (needs ~162 MB) ----
  char* p = (char*)d_ws;
  auto alloc = [&](size_t bytes) -> void* {
    void* r = (void*)p;
    p += (bytes + 255) & ~(size_t)255;
    return r;
  };
  bf16_t* x0   = (bf16_t*)alloc((size_t)B_DIM * SEQ_LEN * 64 * 2);       // padded x, layer0
  bf16_t* lout = (bf16_t*)alloc((size_t)B_DIM * SEQ_LEN * H_DIM * 2);    // layer0 outputs
  bf16_t* W0x  = (bf16_t*)alloc((size_t)2048 * 64 * 2);
  bf16_t* W0h  = (bf16_t*)alloc((size_t)2048 * 512 * 2);
  bf16_t* W1x  = (bf16_t*)alloc((size_t)2048 * 512 * 2);
  bf16_t* W1h  = (bf16_t*)alloc((size_t)2048 * 512 * 2);
  bf16_t* Wd0b = (bf16_t*)alloc((size_t)512 * 512 * 2);
  bf16_t* Wd1b = (bf16_t*)alloc((size_t)512 * 512 * 2);
  bf16_t* hb16 = (bf16_t*)alloc((size_t)512 * 512 * 2);
  bf16_t* hadj = (bf16_t*)alloc((size_t)512 * 512 * 2);
  float*  hf32 = (float*) alloc((size_t)512 * 512 * 4);
  float*  cbuf = (float*) alloc((size_t)512 * 512 * 4);

  auto conv = [&](bf16_t* dst, const float* src, long rows, long sld, long off,
                  int cc, int dc) {
    long total = rows * (long)dc;
    int blk = 256;
    long nb = (total + blk - 1) / blk;
    conv_f32_bf16_kernel<<<dim3((unsigned)nb), blk, 0, stream>>>(
        dst, src, rows, sld, off, cc, dc);
  };
  auto zero = [&](void* dst, size_t bytes) {
    long n4 = (long)(bytes / 16);
    int blk = 256;
    long nb = (n4 + blk - 1) / blk;
    zero_u4_kernel<<<dim3((unsigned)nb), blk, 0, stream>>>((uint4*)dst, n4);
  };

  // ---- one-time (per call) bf16 weight / input prep ----
  conv(x0,   in_t, (long)B_DIM * SEQ_LEN, 64,   0,  63, 64);   // pad col 63 = 0
  conv(W0x,  W0,   2048, 575,  0,   63, 64);                    // pad col 63 = 0
  conv(W0h,  W0,   2048, 575,  63,  512, 512);
  conv(W1x,  W1,   2048, 1024, 0,   512, 512);
  conv(W1h,  W1,   2048, 1024, 512, 512, 512);
  conv(Wd0b, Wd0,  512,  512,  0,   512, 512);
  conv(Wd1b, Wd1,  512,  512,  0,   512, 512);

  const dim3 gridG(16, 8);   // N-tiles(32) x M-tiles(64)
  const int  blk = 128;

  for (int layer = 0; layer < 2; ++layer) {
    zero(hb16, (size_t)512 * 512 * 2);
    zero(hf32, (size_t)512 * 512 * 4);
    zero(cbuf, (size_t)512 * 512 * 4);

    const bf16_t* Wxl = layer ? W1x : W0x;
    const bf16_t* Whl = layer ? W1h : W0h;
    const bf16_t* Wdl = layer ? Wd1b : Wd0b;
    const float*  bl  = layer ? b1 : b0;
    const float*  bdl = layer ? bd1 : bd0;
    const int Kx = layer ? 512 : 64;

    for (int s = 0; s < SEQ_LEN; ++s) {
      const int j = SEQ_LEN - 1 - s;                 // original time index
      const float* DT = in_t + (long)j * 64 + 63;    // dt column

      tlstm_hadj_kernel<<<gridG, blk, 0, stream>>>(
          hb16, Wdl, bdl, hf32, DT, (long)SEQ_LEN * 64, hadj);

      const bf16_t* Ax = layer ? (lout + (long)j * H_DIM)
                               : (x0   + (long)j * 64);
      const long ldax  = layer ? (long)SEQ_LEN * H_DIM : (long)SEQ_LEN * 64;
      bf16_t* Lo  = layer ? (bf16_t*)nullptr : (lout + (long)s * H_DIM);
      float* HfO  = (layer == 1 && s == SEQ_LEN - 1) ? (float*)d_out : hf32;

      tlstm_gates_kernel<<<gridG, blk, 0, stream>>>(
          Ax, ldax, Kx, hadj, Wxl, Whl, bl,
          cbuf, HfO, hb16, Lo, (long)SEQ_LEN * H_DIM);
    }
  }
}
